// STMamba_11699490914655
// MI455X (gfx1250) — matrix operand order repaired and checked by reading, not compiled, not run on hardware
//
#include <hip/hip_runtime.h>
#include <math.h>

// ---------------------------------------------------------------------------
// ST-Mamba forward for MI455X (gfx1250, wave32, WMMA)
//   kernel A: per-image fused conv3d+BN+ReLU (scalar) -> LDS bf16 ->
//             conv2d as implicit GEMM via v_wmma_f32_16x16x32_bf16 ->
//             BN+ReLU+spiral-gather+cls+pos_emb -> tok[B,82,64]
//   kernel B: per-image LN + 3x(64x64) projections via WMMA + squeeze-token +
//             sequential S6 scan -> pooled[B,64]
//   kernel C: LN + MLP (gelu exact) + residual -> out[B,64]
// ---------------------------------------------------------------------------

typedef __attribute__((ext_vector_type(16))) __bf16    v16bf;
typedef __attribute__((ext_vector_type(8)))  float     v8f;
typedef __attribute__((ext_vector_type(4)))  unsigned  u32x4;

union Frag { v16bf v; u32x4 q[2]; };

#define NBATCH 4096
#define LTOK   82
#define NPOS   81
#define NC     64
#define KCH    72          // 9 taps * 8 chunks of 32 (channels padded 240->256)

static __device__ __forceinline__ unsigned short f2bf(float f) {
    unsigned u = __float_as_uint(f);
    unsigned r = u + 0x7FFFu + ((u >> 16) & 1u);   // RNE
    return (unsigned short)(r >> 16);
}
static __device__ __forceinline__ float sigm(float x) { return 1.0f / (1.0f + expf(-x)); }

// ------------------------- prep: spiral inverse map -------------------------
__global__ void prep_spiral(int* __restrict__ inv) {
    if (threadIdx.x == 0 && blockIdx.x == 0) {
        int rows[81], cols[81];
        const int cen = 4;
        int idx = 0;
        rows[0] = cen; cols[0] = cen; idx = 1;
        for (int r = 1; r <= cen; ++r) {
            for (int i = 0; i < 2 * r; ++i) { rows[idx] = cen - r;         cols[idx] = cen - r + 1 + i; ++idx; }
            for (int i = 0; i < 2 * r; ++i) { rows[idx] = cen - r + 1 + i; cols[idx] = cen + r;         ++idx; }
            for (int i = 0; i < 2 * r; ++i) { rows[idx] = cen + r;         cols[idx] = cen - r + i;     ++idx; }
            for (int i = 0; i < 2 * r; ++i) { rows[idx] = cen - r + i;     cols[idx] = cen - r;         ++idx; }
        }
        for (int s = 0; s < 81; ++s) inv[rows[s] * 9 + cols[s]] = s;
    }
}

// ----------- prep: pack conv2d weights into WMMA-B fragment layout ----------
// layout: packC[((kcG*4 + nt)*32 + lane)*16 + e], kcG = tap*8 + kchunk
// B-frag per-lane K map: K = (lane>=16 ? 8 : 0) + (e&7) + (e>=8 ? 16 : 0)  (ISA table order)
__global__ void prep_packC(const float* __restrict__ w2c, unsigned short* __restrict__ packC) {
    int idx = blockIdx.x * 256 + threadIdx.x;
    if (idx >= KCH * 4 * 32 * 16) return;
    int e    = idx & 15;
    int lane = (idx >> 4) & 31;
    int nt   = (idx >> 9) & 3;
    int kcG  = idx >> 11;
    int K    = ((lane >> 4) << 3) + (e & 7) + ((e >> 3) << 4);
    int t    = kcG >> 3;                 // tap = ky*3+kx
    int cpad = (kcG & 7) * 32 + K;       // padded input channel (0..255)
    int n    = nt * 16 + (lane & 15);    // output channel
    float v  = (cpad < 240) ? w2c[n * 2160 + cpad * 9 + t] : 0.0f;  // (64,240,3,3)
    packC[idx] = f2bf(v);
}

// -------- prep: pack Wb|Wc|Wd into one [K=64][N=192] fragment buffer --------
__global__ void prep_packS(const float* __restrict__ Wb, const float* __restrict__ Wc,
                           const float* __restrict__ Wd, unsigned short* __restrict__ packS) {
    int idx = blockIdx.x * 256 + threadIdx.x;
    if (idx >= 2 * 12 * 32 * 16) return;
    int e    = idx & 15;
    int lane = (idx >> 4) & 31;
    int rest = idx >> 9;
    int nt   = rest % 12;
    int kc   = rest / 12;
    int K    = ((lane >> 4) << 3) + (e & 7) + ((e >> 3) << 4);
    int k    = kc * 32 + K;
    int ncol = (nt & 3) * 16 + (lane & 15);
    const float* W = (nt < 4) ? Wb : ((nt < 8) ? Wc : Wd);
    packS[idx] = f2bf(W[k * 64 + ncol]);
}

// ----------------------------- kernel A ------------------------------------
__global__ __launch_bounds__(256)
void kA(const float* __restrict__ in, const float* __restrict__ w3,
        const float* __restrict__ b3, const float* __restrict__ g3,
        const float* __restrict__ be3, const float* __restrict__ m3, const float* __restrict__ v3,
        const float* __restrict__ b2c, const float* __restrict__ g2,
        const float* __restrict__ be2, const float* __restrict__ m2, const float* __restrict__ v2,
        const float* __restrict__ pos_emb, const float* __restrict__ cls_tok,
        const unsigned short* __restrict__ packC, const int* __restrict__ invsp,
        float* __restrict__ tok)
{
    __shared__ __align__(16) float          s_x[32 * 121];        // conv3d padded input
    __shared__ __align__(16) unsigned short s_h[121 * 256];       // conv2d padded bf16 activations
    __shared__ float s_w3[216];
    __shared__ float s_a3[8], s_c3[8], s_a2[64], s_c2[64];
    __shared__ int   s_inv[81];

    const int tid = threadIdx.x;
    const int b   = blockIdx.x;

    for (int i = tid; i < 32 * 121; i += 256) s_x[i] = 0.0f;
    for (int i = tid; i < (121 * 256) / 2; i += 256) ((unsigned*)s_h)[i] = 0u;
    if (tid < 216) s_w3[tid] = w3[tid];
    if (tid < 8) {
        float a = g3[tid] * rsqrtf(v3[tid] + 1e-5f);
        s_a3[tid] = a;
        s_c3[tid] = (b3[tid] - m3[tid]) * a + be3[tid];
    }
    if (tid >= 32 && tid < 96) {
        int n = tid - 32;
        float a = g2[n] * rsqrtf(v2[n] + 1e-5f);
        s_a2[n] = a;
        s_c2[n] = (b2c[n] - m2[n]) * a + be2[n];
    }
    if (tid >= 96 && tid < 177) s_inv[tid - 96] = invsp[tid - 96];
    __syncthreads();

    // load input [30,9,9] into zero-padded [32,11,11]
    const float* xb = in + (size_t)b * 2430;
    for (int i = tid; i < 2430; i += 256) {
        int d = i / 81, pos = i % 81;
        s_x[(d + 1) * 121 + (pos / 9 + 1) * 11 + (pos % 9 + 1)] = xb[i];
    }
    __syncthreads();

    // conv3d + BN + ReLU -> bf16 LDS, channel = o*30 + d, padded 11x11 grid
    for (int i = tid; i < 240 * 81; i += 256) {
        int ch = i / 81, pos = i % 81;
        int o = ch / 30, d = ch % 30;
        int r = pos / 9, c = pos % 9;
        float acc = 0.0f;
        const float* w = &s_w3[o * 27];
        #pragma unroll
        for (int dz = 0; dz < 3; ++dz)
            #pragma unroll
            for (int dy = 0; dy < 3; ++dy)
                #pragma unroll
                for (int dx = 0; dx < 3; ++dx)
                    acc += s_x[(d + dz) * 121 + (r + dy) * 11 + (c + dx)] * w[dz * 9 + dy * 3 + dx];
        float val = acc * s_a3[o] + s_c3[o];
        val = val > 0.0f ? val : 0.0f;
        s_h[((r + 1) * 11 + (c + 1)) * 256 + ch] = f2bf(val);
    }
    __syncthreads();

    // conv2d implicit GEMM: [96(81) x 2304] * [2304 x 64] via WMMA bf16
    const int wave = tid >> 5, lane = tid & 31;
    const int half = lane >> 4, nlo = lane & 15;
    const int ntw = wave & 3;         // N tile 0..3
    const int mt0 = wave >> 2;        // M tiles {mt0, mt0+2, mt0+4}

    v8f zero = {0.f, 0.f, 0.f, 0.f, 0.f, 0.f, 0.f, 0.f};
    v8f accs[3];
    accs[0] = zero; accs[1] = zero; accs[2] = zero;

    int sp[3];
    #pragma unroll
    for (int i = 0; i < 3; ++i) {
        int m = (mt0 + 2 * i) * 16 + nlo;     // A-row (M) = lane&15 within tile
        if (m >= 81) m = 0;                    // dummy rows, discarded on store
        sp[i] = (m / 9 + 1) * 11 + (m % 9 + 1);
    }

    for (int kcG = 0; kcG < KCH; ++kcG) {
        int t   = kcG >> 3, kcl = kcG & 7;
        int dy  = t / 3 - 1, dx = t % 3 - 1;
        Frag bf;
        const u32x4* bp = reinterpret_cast<const u32x4*>(packC + (((kcG * 4 + ntw) * 32 + lane) << 4));
        bf.q[0] = bp[0];
        bf.q[1] = bp[1];
        int cb = kcl * 32 + half * 8;         // A-frag K layout (ISA 16-bit A map)
        #pragma unroll
        for (int i = 0; i < 3; ++i) {
            Frag af;
            const unsigned short* arow = &s_h[(sp[i] + dy * 11 + dx) * 256 + cb];
            af.q[0] = *reinterpret_cast<const u32x4*>(arow);
            af.q[1] = *reinterpret_cast<const u32x4*>(arow + 16);
            accs[i] = __builtin_amdgcn_wmma_f32_16x16x32_bf16(
                false, af.v, false, bf.v, (short)0, accs[i], false, false);
        }
    }

    // epilogue: BN + ReLU + spiral gather + pos_emb -> tok
    float* tokb = tok + (size_t)b * (LTOK * NC);
    const int n = ntw * 16 + nlo;
    const float a2 = s_a2[n], c2 = s_c2[n];
    #pragma unroll
    for (int i = 0; i < 3; ++i) {
        int mt = mt0 + 2 * i;
        #pragma unroll
        for (int r8 = 0; r8 < 8; ++r8) {
            int m = mt * 16 + r8 + half * 8;   // D layout: M = vgpr + 8*half
            if (m < 81) {
                float val = accs[i][r8] * a2 + c2;
                val = val > 0.0f ? val : 0.0f;
                int tix = s_inv[m] + 1;
                tokb[tix * NC + n] = val + pos_emb[tix * NC + n];
            }
        }
    }
    if (tid < 64) tokb[tid] = cls_tok[tid] + pos_emb[tid];   // cls token
}

// ----------------------------- kernel B ------------------------------------
__global__ __launch_bounds__(256)
void kB(const float* __restrict__ tok,
        const float* __restrict__ ln1_g, const float* __restrict__ ln1_b,
        const float* __restrict__ bb, const float* __restrict__ bc, const float* __restrict__ bd,
        const float* __restrict__ delta_p, const float* __restrict__ A_p,
        const float* __restrict__ Wst, const float* __restrict__ bst,
        const float* __restrict__ bias_st,
        const unsigned short* __restrict__ packS,
        float* __restrict__ pooled)
{
    __shared__ float s_tok[LTOK * NC];
    __shared__ __align__(16) unsigned short s_xnb[96 * NC];   // bf16 xn, rows 82..95 zero
    __shared__ float s_mm[LTOK * 192];                        // [B0 | C | D] raw matmuls
    __shared__ float s_mu[LTOK], s_rs[LTOK];
    __shared__ float s_sq[NC], s_sq2[NC], s_g[NC], s_b[NC];

    const int tid = threadIdx.x;
    const int b   = blockIdx.x;
    const float* tb = tok + (size_t)b * (LTOK * NC);

    for (int i = tid; i < LTOK * NC; i += 256) s_tok[i] = tb[i];
    if (tid < 64) { s_g[tid] = ln1_g[tid]; s_b[tid] = ln1_b[tid]; }
    __syncthreads();

    if (tid < LTOK) {
        float m = 0.0f;
        for (int d = 0; d < NC; ++d) m += s_tok[tid * NC + d];
        m *= (1.0f / NC);
        float v = 0.0f;
        for (int d = 0; d < NC; ++d) { float t = s_tok[tid * NC + d] - m; v += t * t; }
        v *= (1.0f / NC);
        s_mu[tid] = m;
        s_rs[tid] = rsqrtf(v + 1e-5f);
    }
    __syncthreads();

    for (int i = tid; i < 96 * NC; i += 256) {
        if (i < LTOK * NC) {
            int l = i >> 6, d = i & 63;
            float xn = (s_tok[i] - s_mu[l]) * s_rs[l] * s_g[d] + s_b[d];
            s_xnb[i] = f2bf(xn);
        } else {
            s_xnb[i] = 0;
        }
    }
    if (tid < 64) {       // squeeze token: mean over tokens of xn
        float g = s_g[tid], bo = s_b[tid], sum = 0.0f;
        for (int l = 0; l < LTOK; ++l)
            sum += (s_tok[l * NC + tid] - s_mu[l]) * s_rs[l] * g + bo;
        s_sq[tid] = sum / (float)LTOK;
    }
    __syncthreads();

    // [96x64] x [64x192] (= xn @ [Wb|Wc|Wd]) via WMMA
    const int wave = tid >> 5, lane = tid & 31;
    const int half = lane >> 4, nlo = lane & 15;
    v8f zero = {0.f, 0.f, 0.f, 0.f, 0.f, 0.f, 0.f, 0.f};
    for (int i = 0; i < 9; ++i) {
        int tile = wave * 9 + i;      // 72 tiles = 6 M x 12 N
        int mt = tile / 12, nt = tile % 12;
        int m = mt * 16 + nlo;
        v8f acc = zero;
        #pragma unroll
        for (int kc = 0; kc < 2; ++kc) {
            Frag af, bf;
            int cb = kc * 32 + half * 8;
            af.q[0] = *reinterpret_cast<const u32x4*>(&s_xnb[m * NC + cb]);
            af.q[1] = *reinterpret_cast<const u32x4*>(&s_xnb[m * NC + cb + 16]);
            const u32x4* bp = reinterpret_cast<const u32x4*>(packS + (((kc * 12 + nt) * 32 + lane) << 4));
            bf.q[0] = bp[0];
            bf.q[1] = bp[1];
            acc = __builtin_amdgcn_wmma_f32_16x16x32_bf16(
                false, af.v, false, bf.v, (short)0, acc, false, false);
        }
        int col = nt * 16 + nlo;
        #pragma unroll
        for (int r8 = 0; r8 < 8; ++r8) {
            int mm = mt * 16 + r8 + half * 8;
            if (mm < LTOK) s_mm[mm * 192 + col] = acc[r8];
        }
    }
    __syncthreads();

    if (tid < 64) {       // sq2 = sigmoid(relu(sq @ Wst + bst)) + bias_st
        float a = bst[tid];
        for (int k = 0; k < 64; ++k) a += s_sq[k] * Wst[k * 64 + tid];
        a = a > 0.0f ? a : 0.0f;
        s_sq2[tid] = sigm(a) + bias_st[tid];
    }
    __syncthreads();

    if (tid < 64) {       // selective scan + residual + pool, one channel per lane
        const int d = tid;
        const float g = s_g[d], bo = s_b[d];
        const float bbv = bb[d], bcv = bc[d], bdv = bd[d];
        const float t2 = s_sq2[d];
        float s = 0.0f, accum = 0.0f;
        for (int l = 0; l < LTOK; ++l) {
            float tokv = s_tok[l * NC + d];
            float xn = (tokv - s_mu[l]) * s_rs[l] * g + bo;
            float b0 = s_mm[l * 192 + d] + bbv;
            float c  = s_mm[l * 192 + 64 + d] + bcv;
            float dl = sigm(s_mm[l * 192 + 128 + d] + bdv + delta_p[l * NC + d]);
            float a  = dl * A_p[l * NC + d];
            float bx = dl * b0 * xn;
            s = a * s + bx;
            float out = c * s + t2 * xn;
            accum += tokv + out * sigm(xn);
        }
        pooled[(size_t)b * NC + d] = accum / (float)LTOK;
    }
}

// ----------------------------- kernel C ------------------------------------
__global__ __launch_bounds__(64)
void kC(const float* __restrict__ pooled,
        const float* __restrict__ ln2_g, const float* __restrict__ ln2_b,
        const float* __restrict__ Wm1, const float* __restrict__ bm1,
        const float* __restrict__ Wm2, const float* __restrict__ bm2,
        float* __restrict__ out)
{
    __shared__ float s_p[64], s_y[64], s_h[64];
    const int t = threadIdx.x;
    const int b = blockIdx.x;
    float p = pooled[(size_t)b * 64 + t];
    s_p[t] = p;
    __syncthreads();
    float m = 0.0f;
    for (int k = 0; k < 64; ++k) m += s_p[k];
    m *= (1.0f / 64.0f);
    float v = 0.0f;
    for (int k = 0; k < 64; ++k) { float d = s_p[k] - m; v += d * d; }
    v *= (1.0f / 64.0f);
    s_y[t] = (p - m) * rsqrtf(v + 1e-5f) * ln2_g[t] + ln2_b[t];
    __syncthreads();
    float h = bm1[t];
    for (int k = 0; k < 64; ++k) h += s_y[k] * Wm1[k * 64 + t];
    s_h[t] = 0.5f * h * (1.0f + erff(h * 0.70710678118654752f));   // exact gelu
    __syncthreads();
    float o = bm2[t];
    for (int k = 0; k < 64; ++k) o += s_h[k] * Wm2[k * 64 + t];
    out[(size_t)b * 64 + t] = o + p;
}

// ----------------------------- launcher ------------------------------------
extern "C" void kernel_launch(void* const* d_in, const int* in_sizes, int n_in,
                              void* d_out, int out_size, void* d_ws, size_t ws_size,
                              hipStream_t stream)
{
    (void)in_sizes; (void)n_in; (void)out_size;
    const float* in_data = (const float*)d_in[0];
    const float* w3      = (const float*)d_in[1];
    const float* b3      = (const float*)d_in[2];
    const float* g3      = (const float*)d_in[3];
    const float* be3     = (const float*)d_in[4];
    const float* m3      = (const float*)d_in[5];
    const float* v3      = (const float*)d_in[6];
    const float* w2c     = (const float*)d_in[7];
    const float* b2c     = (const float*)d_in[8];
    const float* g2      = (const float*)d_in[9];
    const float* be2     = (const float*)d_in[10];
    const float* m2      = (const float*)d_in[11];
    const float* v2      = (const float*)d_in[12];
    const float* pos_emb = (const float*)d_in[13];
    const float* cls_tok = (const float*)d_in[14];
    const float* ln1_g   = (const float*)d_in[15];
    const float* ln1_b   = (const float*)d_in[16];
    const float* Wb      = (const float*)d_in[17];
    const float* bb      = (const float*)d_in[18];
    const float* Wc      = (const float*)d_in[19];
    const float* bc      = (const float*)d_in[20];
    const float* Wd      = (const float*)d_in[21];
    const float* bd      = (const float*)d_in[22];
    const float* delta_p = (const float*)d_in[23];
    const float* A_p     = (const float*)d_in[24];
    const float* Wst     = (const float*)d_in[25];
    const float* bst     = (const float*)d_in[26];
    const float* bias_st = (const float*)d_in[27];
    const float* ln2_g   = (const float*)d_in[28];
    const float* ln2_b   = (const float*)d_in[29];
    const float* Wm1     = (const float*)d_in[30];
    const float* bm1     = (const float*)d_in[31];
    const float* Wm2     = (const float*)d_in[32];
    const float* bm2     = (const float*)d_in[33];

    // workspace layout
    const size_t OFF_INV   = 0;                          // 81 ints
    const size_t OFF_PACKC = 512;                        // 294912 B
    const size_t OFF_PACKS = 512 + 294912;               // 24576 B
    const size_t OFF_TOK   = 320000;                     // 4096*82*64*4 B
    const size_t OFF_POOL  = OFF_TOK + (size_t)NBATCH * LTOK * NC * 4;
    const size_t NEED      = OFF_POOL + (size_t)NBATCH * NC * 4;
    if (ws_size < NEED) return;

    char* ws = (char*)d_ws;
    int*            inv    = (int*)(ws + OFF_INV);
    unsigned short* packC  = (unsigned short*)(ws + OFF_PACKC);
    unsigned short* packS  = (unsigned short*)(ws + OFF_PACKS);
    float*          tok    = (float*)(ws + OFF_TOK);
    float*          pooled = (float*)(ws + OFF_POOL);

    prep_spiral<<<1, 1, 0, stream>>>(inv);
    prep_packC<<<(KCH * 4 * 32 * 16 + 255) / 256, 256, 0, stream>>>(w2c, packC);
    prep_packS<<<(2 * 12 * 32 * 16 + 255) / 256, 256, 0, stream>>>(Wb, Wc, Wd, packS);

    kA<<<NBATCH, 256, 0, stream>>>(in_data, w3, b3, g3, be3, m3, v3,
                                   b2c, g2, be2, m2, v2,
                                   pos_emb, cls_tok, packC, inv, tok);
    kB<<<NBATCH, 256, 0, stream>>>(tok, ln1_g, ln1_b, bb, bc, bd,
                                   delta_p, A_p, Wst, bst, bias_st, packS, pooled);
    kC<<<NBATCH, 64, 0, stream>>>(pooled, ln2_g, ln2_b, Wm1, bm1, Wm2, bm2,
                                  (float*)d_out);
}
